// MoELayer_2645699854602
// MI455X (gfx1250) — matrix operand order repaired and checked by reading, not compiled, not run on hardware
//
#include <hip/hip_runtime.h>
#include <hip/hip_bf16.h>

// ---------------- problem constants ----------------
#define D_MODEL 1024
#define D_FF    4096
#define NEXP    8
#define TOPK    2
#define NTOK    4096            // BATCH * SEQ = 2 * 2048

// d_out layout (flat, in reference return order, all as float)
#define OUT_Y      0                          // [NTOK, D_MODEL]
#define OUT_LOGITS (NTOK * D_MODEL)           // [NTOK, 8]
#define OUT_IDX    (OUT_LOGITS + NTOK * NEXP) // [NTOK, 2] (indices stored as float)
#define OUT_W      (OUT_IDX + NTOK * TOPK)    // [NTOK, 2]

// workspace layout
#define WS_CNT 0                               // int cnt[8]
#define WS_TOK 256                             // int  btok[8][NTOK]
#define WS_BW  (256 + NEXP * NTOK * 4)         // float bw [8][NTOK]

// ---------------- WMMA types ----------------
typedef __attribute__((ext_vector_type(16))) __bf16 v16bf;
typedef __attribute__((ext_vector_type(8)))  float  v8f;

// hardware f32 -> bf16 conversion (lowers to v_cvt_*bf16_f32 on gfx1250)
__device__ __forceinline__ unsigned short f2bf(float f) {
    __bf16 b = (__bf16)f;
    unsigned short u; __builtin_memcpy(&u, &b, 2); return u;
}
__device__ __forceinline__ __bf16 bfbits(unsigned short u) {
    __bf16 b; __builtin_memcpy(&b, &u, 2); return b;
}

// ---------------- init: zero output + bucket counters ----------------
__global__ void moe_init(float* __restrict__ out, int n, int* __restrict__ cnt) {
    int i = blockIdx.x * blockDim.x + threadIdx.x;
    if (i < n) out[i] = 0.0f;
    if (blockIdx.x == 0 && threadIdx.x < NEXP) cnt[threadIdx.x] = 0;
}

// ---------------- router: logits, softmax, top-2, bucket build ----------------
// one wave32 per token
__global__ __launch_bounds__(256) void moe_router(
    const float* __restrict__ x, const float* __restrict__ gw,
    float* __restrict__ out, int* __restrict__ cnt,
    int* __restrict__ btok, float* __restrict__ bw)
{
    const int wave = threadIdx.x >> 5;
    const int lane = threadIdx.x & 31;
    const int t    = blockIdx.x * 8 + wave;
    if (t >= NTOK) return;

    const float* xr = x + (size_t)t * D_MODEL;
    float acc[NEXP];
#pragma unroll
    for (int e = 0; e < NEXP; ++e) acc[e] = 0.0f;

    for (int k = lane; k < D_MODEL; k += 32) {
        const float xv = xr[k];
        const float4 g0 = *(const float4*)(gw + (size_t)k * NEXP);
        const float4 g1 = *(const float4*)(gw + (size_t)k * NEXP + 4);
        acc[0] += xv * g0.x; acc[1] += xv * g0.y;
        acc[2] += xv * g0.z; acc[3] += xv * g0.w;
        acc[4] += xv * g1.x; acc[5] += xv * g1.y;
        acc[6] += xv * g1.z; acc[7] += xv * g1.w;
    }
#pragma unroll
    for (int e = 0; e < NEXP; ++e) {
        float v = acc[e];
#pragma unroll
        for (int off = 16; off > 0; off >>= 1) v += __shfl_xor(v, off, 32);
        acc[e] = v;                            // all lanes hold the logit
    }

    // softmax over 8 logits (redundantly on all lanes)
    float mx = acc[0];
#pragma unroll
    for (int e = 1; e < NEXP; ++e) mx = fmaxf(mx, acc[e]);
    float p[NEXP], s = 0.0f;
#pragma unroll
    for (int e = 0; e < NEXP; ++e) { p[e] = __expf(acc[e] - mx); s += p[e]; }
    const float inv = 1.0f / s;
#pragma unroll
    for (int e = 0; e < NEXP; ++e) p[e] *= inv;

    // top-2 (lowest index wins ties, matching lax.top_k)
    int i0 = 0; float p0 = p[0];
#pragma unroll
    for (int e = 1; e < NEXP; ++e) if (p[e] > p0) { p0 = p[e]; i0 = e; }
    int i1 = (i0 == 0) ? 1 : 0; float p1 = p[i1];
#pragma unroll
    for (int e = 0; e < NEXP; ++e)
        if (e != i0 && p[e] > p1) { p1 = p[e]; i1 = e; }
    const float rs = 1.0f / (p0 + p1);
    const float w0 = p0 * rs, w1 = p1 * rs;

    if (lane == 0) {
        float* lg = out + OUT_LOGITS + (size_t)t * NEXP;
#pragma unroll
        for (int e = 0; e < NEXP; ++e) lg[e] = acc[e];
        out[OUT_IDX + t * TOPK + 0] = (float)i0;
        out[OUT_IDX + t * TOPK + 1] = (float)i1;
        out[OUT_W   + t * TOPK + 0] = w0;
        out[OUT_W   + t * TOPK + 1] = w1;
        int pos0 = atomicAdd(&cnt[i0], 1);
        btok[i0 * NTOK + pos0] = t;  bw[i0 * NTOK + pos0] = w0;
        int pos1 = atomicAdd(&cnt[i1], 1);
        btok[i1 * NTOK + pos1] = t;  bw[i1 * NTOK + pos1] = w1;
    }
}

// ---------------- fused expert FFN via WMMA bf16 ----------------
#define TILE_M   16
#define FCHUNK   128
#define KSTEP    32
#define NSLICE   256
#define XS_STRIDE 1032   // 1024 + 8 (bank-skew); byte stride 2064 (8B aligned)
#define HS_STRIDE 136    // 128 + 8
#define BS_STRIDE 34     // 32 + 2

__global__ __launch_bounds__(256, 1) void moe_ffn(
    const float* __restrict__ x,
    const float* __restrict__ w1, const float* __restrict__ b1,
    const float* __restrict__ w2, const float* __restrict__ b2,
    const int* __restrict__ cnt, const int* __restrict__ btok,
    const float* __restrict__ bw, float* __restrict__ out)
{
    const int e    = blockIdx.y;
    const int n    = cnt[e];
    const int base = blockIdx.x * TILE_M;
    if (base >= n) return;                       // uniform early-exit (no barriers yet)
    const int rows = min(TILE_M, n - base);

    __shared__ unsigned short xs[TILE_M * XS_STRIDE];   // gathered x tile, bf16
    __shared__ unsigned short hs[TILE_M * HS_STRIDE];   // h chunk, bf16
    __shared__ unsigned short bs[NSLICE * BS_STRIDE];   // transposed B staging, bf16
    __shared__ int   tid_s[TILE_M];
    __shared__ float tw_s[TILE_M];

    if (threadIdx.x < TILE_M) {
        const int m = threadIdx.x;
        if (m < rows) { tid_s[m] = btok[e * NTOK + base + m];
                        tw_s[m]  = bw  [e * NTOK + base + m]; }
        else          { tid_s[m] = 0; tw_s[m] = 0.0f; }
    }
    __syncthreads();

    // gather x rows -> LDS bf16 (b128 global loads, packed ds_store_b64)
    for (int i = threadIdx.x; i < (TILE_M * D_MODEL) / 4; i += 256) {
        const int idx = i * 4;
        const int m = idx >> 10, c = idx & (D_MODEL - 1);
        float4 v = make_float4(0.f, 0.f, 0.f, 0.f);
        if (m < rows) v = *(const float4*)(x + (size_t)tid_s[m] * D_MODEL + c);
        union { unsigned short h[4]; uint2 q; } pk;
        pk.h[0] = f2bf(v.x); pk.h[1] = f2bf(v.y);
        pk.h[2] = f2bf(v.z); pk.h[3] = f2bf(v.w);
        *(uint2*)&xs[m * XS_STRIDE + c] = pk.q;          // 8B-aligned
    }

    const int wave  = threadIdx.x >> 5;
    const int lane  = threadIdx.x & 31;
    const int lhalf = lane >> 4;
    const int lidx  = lane & 15;

    const float* w1e = w1 + (size_t)e * D_MODEL * D_FF;
    const float* w2e = w2 + (size_t)e * D_FF * D_MODEL;

    v8f yacc[8];
#pragma unroll
    for (int q = 0; q < 8; ++q) yacc[q] = {};

    for (int fc = 0; fc < D_FF / FCHUNK; ++fc) {
        // ================= GEMM1: h = relu(x @ w1[:, chunk] + b1) =================
        v8f c1 = {};
        for (int k0 = 0; k0 < D_MODEL; k0 += KSTEP) {
            __syncthreads();                                   // bs safe to overwrite
            if (k0 + KSTEP < D_MODEL)                          // gfx1250 prefetch path
                __builtin_prefetch(w1e + (size_t)(k0 + KSTEP) * D_FF + fc * FCHUNK, 0, 1);
            // stage w1[k0:k0+32, chunk] -> bs[col][k] (bf16, transposed)
            for (int i = threadIdx.x; i < (KSTEP * FCHUNK) / 4; i += 256) {
                const int idx = i * 4;
                const int r = idx / FCHUNK, c = idx % FCHUNK;
                const float4 v = *(const float4*)(w1e + (size_t)(k0 + r) * D_FF
                                                  + fc * FCHUNK + c);
                bs[(c + 0) * BS_STRIDE + r] = f2bf(v.x);
                bs[(c + 1) * BS_STRIDE + r] = f2bf(v.y);
                bs[(c + 2) * BS_STRIDE + r] = f2bf(v.z);
                bs[(c + 3) * BS_STRIDE + r] = f2bf(v.w);
            }
            __syncthreads();

            // A fragment (K = 16*half .. +16 consecutive, matching B)
            const unsigned short* xrow = &xs[lidx * XS_STRIDE];
            v16bf a;
#pragma unroll
            for (int v = 0; v < 8; ++v) {
                const int kk = k0 + 16 * lhalf + 2 * v;
                a[2 * v]     = bfbits(xrow[kk]);
                a[2 * v + 1] = bfbits(xrow[kk + 1]);
            }
            // B fragment (N = lane%16, K = 16*half .. +16 consecutive)
            const unsigned short* brow = &bs[(wave * 16 + lidx) * BS_STRIDE + 16 * lhalf];
            v16bf b;
#pragma unroll
            for (int v = 0; v < 16; ++v) b[v] = bfbits(brow[v]);

            c1 = __builtin_amdgcn_wmma_f32_16x16x32_bf16(
                    false, a, false, b, (short)0, c1, false, false);
        }

        // bias + relu + write h chunk (bf16) to LDS
        const int hcol = wave * 16 + lidx;
        const float bias1 = b1[(size_t)e * D_FF + fc * FCHUNK + hcol];
        __syncthreads();                                       // hs safe to overwrite
#pragma unroll
        for (int r = 0; r < 8; ++r) {
            const int m = r + 8 * lhalf;
            hs[m * HS_STRIDE + hcol] = f2bf(fmaxf(c1[r] + bias1, 0.0f));
        }
        __syncthreads();

        // ================= GEMM2: y += h(16x128) @ w2[chunk, :] =================
        const int f0 = fc * FCHUNK;
        for (int k0 = 0; k0 < FCHUNK; k0 += KSTEP) {           // 4 K-steps (A hoisted)
            // A fragment from h chunk — shared across all 4 N-slices
            const unsigned short* hrow = &hs[lidx * HS_STRIDE];
            v16bf a;
#pragma unroll
            for (int v = 0; v < 8; ++v) {
                const int kk = k0 + 16 * lhalf + 2 * v;
                a[2 * v]     = bfbits(hrow[kk]);
                a[2 * v + 1] = bfbits(hrow[kk + 1]);
            }
            for (int s = 0; s < 4; ++s) {                      // 4 N-slices of 256
                __syncthreads();
                for (int i = threadIdx.x; i < (KSTEP * NSLICE) / 4; i += 256) {
                    const int idx = i * 4;
                    const int r = idx / NSLICE, c = idx % NSLICE;
                    const float4 v = *(const float4*)(w2e + (size_t)(f0 + k0 + r) * D_MODEL
                                                      + s * NSLICE + c);
                    bs[(c + 0) * BS_STRIDE + r] = f2bf(v.x);
                    bs[(c + 1) * BS_STRIDE + r] = f2bf(v.y);
                    bs[(c + 2) * BS_STRIDE + r] = f2bf(v.z);
                    bs[(c + 3) * BS_STRIDE + r] = f2bf(v.w);
                }
                __syncthreads();

#pragma unroll
                for (int j = 0; j < 2; ++j) {                  // 2 N-subtiles per wave
                    const unsigned short* brow =
                        &bs[(wave * 32 + j * 16 + lidx) * BS_STRIDE + 16 * lhalf];
                    v16bf b;
#pragma unroll
                    for (int v = 0; v < 16; ++v) b[v] = bfbits(brow[v]);
                    yacc[s * 2 + j] = __builtin_amdgcn_wmma_f32_16x16x32_bf16(
                        false, a, false, b, (short)0, yacc[s * 2 + j], false, false);
                }
            }
        }
    }

    // ---- scatter: out[token] += weight * (y + b2)  (f32 atomics) ----
#pragma unroll
    for (int s = 0; s < 4; ++s) {
#pragma unroll
        for (int j = 0; j < 2; ++j) {
            const int col = s * NSLICE + wave * 32 + j * 16 + lidx;
            const float bias2 = b2[(size_t)e * D_MODEL + col];
            const v8f yv = yacc[s * 2 + j];
#pragma unroll
            for (int r = 0; r < 8; ++r) {
                const int m = r + 8 * lhalf;
                if (m < rows) {
                    atomicAdd(out + (size_t)tid_s[m] * D_MODEL + col,
                              (yv[r] + bias2) * tw_s[m]);
                }
            }
        }
    }
}

// ---------------- host launcher ----------------
extern "C" void kernel_launch(void* const* d_in, const int* in_sizes, int n_in,
                              void* d_out, int out_size, void* d_ws, size_t ws_size,
                              hipStream_t stream) {
    (void)in_sizes; (void)n_in; (void)ws_size;
    const float* x   = (const float*)d_in[0];
    const float* gw  = (const float*)d_in[1];
    const float* w1  = (const float*)d_in[2];
    const float* b1  = (const float*)d_in[3];
    const float* w2  = (const float*)d_in[4];
    const float* b2  = (const float*)d_in[5];
    float* out = (float*)d_out;
    char*  ws  = (char*)d_ws;

    int*   cnt  = (int*)(ws + WS_CNT);
    int*   btok = (int*)(ws + WS_TOK);
    float* bwp  = (float*)(ws + WS_BW);

    moe_init<<<(out_size + 255) / 256, 256, 0, stream>>>(out, out_size, cnt);
    moe_router<<<NTOK / 8, 256, 0, stream>>>(x, gw, out, cnt, btok, bwp);

    dim3 grid(NTOK / TILE_M, NEXP);
    moe_ffn<<<grid, 256, 0, stream>>>(x, w1, b1, w2, b2, cnt, btok, bwp, out);
}
